// MoE_66159676227785
// MI455X (gfx1250) — matrix operand order repaired and checked by reading, not compiled, not run on hardware
//
#include <hip/hip_runtime.h>

// Problem constants (from the reference).
#define NTOK   8192
#define DMODEL 1024
#define NEXP   8
#define DFFDIM 4096
#define TILE_M 32                      // gathered tokens per workgroup
#define NFCH   64                      // DFF chunk per fused iteration
#define TILES_PER_E (NTOK / TILE_M)    // 256 (upper bound: all tokens -> one expert)

typedef __attribute__((ext_vector_type(16))) __bf16 v16bf;
typedef __attribute__((ext_vector_type(8)))  float  v8f;

union FragU { v16bf v; uint4 q[2]; unsigned short u[16]; };

static __device__ __forceinline__ unsigned short f32_to_bf16(float f) {
  unsigned int u = __float_as_uint(f);
  u += 0x7FFFu + ((u >> 16) & 1u);     // round-to-nearest-even
  return (unsigned short)(u >> 16);
}

static __device__ __forceinline__ v8f wmma_bf16(v16bf a, v16bf b, v8f c) {
  // v_wmma_f32_16x16x32_bf16  D = A(16x32) * B(32x16) + C
  return __builtin_amdgcn_wmma_f32_16x16x32_bf16(false, a, false, b,
                                                 (short)0, c, false, false);
}

// A fragment (16x32 bf16) from a per-lane row pointer (k contiguous in memory).
// ISA layout: row = lane%16; k = {8*half + 0..7, 16 + 8*half + 0..7}
// => two contiguous 16B runs per lane -> two b128 loads.
static __device__ __forceinline__ v16bf load_A_row(const unsigned short* row, int k0) {
  int lane = threadIdx.x & 31;
  int hl   = lane >> 4;
  FragU f;
  f.q[0] = *(const uint4*)(row + k0 + 8 * hl);
  f.q[1] = *(const uint4*)(row + k0 + 16 + 8 * hl);
  return f.v;
}

// B fragment (32x16 bf16) from an n-major (transposed) bf16 matrix:
// ISA layout: col = lane%16; k = {8*half + 0..7, 16 + 8*half + 0..7}
// => two contiguous 16B runs per lane -> two b128 loads.
static __device__ __forceinline__ v16bf load_B_nmajor(const unsigned short* g, int ld,
                                                      int k0, int n0) {
  int lane = threadIdx.x & 31;
  int hl   = lane >> 4;
  const unsigned short* p = g + (size_t)(n0 + (lane & 15)) * ld + k0 + 8 * hl;
  FragU f;
  f.q[0] = *(const uint4*)(p);
  f.q[1] = *(const uint4*)(p + 16);
  return f.v;
}

// ---------------------------------------------------------------------------
// One-time f32 -> bf16 convert (row-major preserved), 8 elements/thread.
// ---------------------------------------------------------------------------
__global__ __launch_bounds__(256)
void convert_bf16_kernel(const float* __restrict__ src, unsigned short* __restrict__ dst) {
  size_t i = ((size_t)blockIdx.x * 256 + threadIdx.x) * 8;
  float4 f0 = *(const float4*)(src + i);
  float4 f1 = *(const float4*)(src + i + 4);
  uint4 o;
  o.x = f32_to_bf16(f0.x) | ((unsigned)f32_to_bf16(f0.y) << 16);
  o.y = f32_to_bf16(f0.z) | ((unsigned)f32_to_bf16(f0.w) << 16);
  o.z = f32_to_bf16(f1.x) | ((unsigned)f32_to_bf16(f1.y) << 16);
  o.w = f32_to_bf16(f1.z) | ((unsigned)f32_to_bf16(f1.w) << 16);
  *(uint4*)(dst + i) = o;
}

// ---------------------------------------------------------------------------
// One-time f32 -> bf16 convert + transpose: src is [batch][R][C] row-major f32,
// dst is [batch][C][R] row-major bf16.  32x32 tiles via LDS, block = 32x8.
// ---------------------------------------------------------------------------
__global__ __launch_bounds__(256)
void transpose_bf16_kernel(const float* __restrict__ src, unsigned short* __restrict__ dst,
                           int R, int C) {
  __shared__ unsigned short tile[32][33];
  size_t base = (size_t)blockIdx.z * (size_t)R * (size_t)C;
  src += base; dst += base;
  int rb = blockIdx.y * 32, cb = blockIdx.x * 32;
  int tx = threadIdx.x & 31, ty = threadIdx.x >> 5;
#pragma unroll
  for (int i = 0; i < 4; ++i) {
    int r = ty + i * 8;
    tile[r][tx] = f32_to_bf16(src[(size_t)(rb + r) * C + cb + tx]);
  }
  __syncthreads();
#pragma unroll
  for (int i = 0; i < 4; ++i) {
    int c = ty + i * 8;
    dst[(size_t)(cb + c) * R + rb + tx] = tile[tx][c];
  }
}

// ---------------------------------------------------------------------------
// Router: f32 logits, softmax, top-2, renormalize, scatter to per-expert lists.
// One wave per token; router stays f32 scalar (top-k is a discrete decision).
// ---------------------------------------------------------------------------
__global__ __launch_bounds__(256)
void moe_router_kernel(const float* __restrict__ x, const float* __restrict__ gw,
                       int* __restrict__ counts, int* __restrict__ toks,
                       float* __restrict__ wts) {
  int wave = threadIdx.x >> 5;
  int lane = threadIdx.x & 31;
  int n    = blockIdx.x * 8 + wave;

  float acc[NEXP];
#pragma unroll
  for (int e = 0; e < NEXP; ++e) acc[e] = 0.0f;

  const float* xr = x + (size_t)n * DMODEL;
  for (int d = lane; d < DMODEL; d += 32) {
    float xv = xr[d];
#pragma unroll
    for (int e = 0; e < NEXP; ++e) acc[e] += xv * gw[d * NEXP + e];
  }
#pragma unroll
  for (int off = 16; off > 0; off >>= 1) {
#pragma unroll
    for (int e = 0; e < NEXP; ++e) acc[e] += __shfl_down(acc[e], off, 32);
  }

  if (lane == 0) {
    float b0 = -3.4e38f; int i0 = 0;
#pragma unroll
    for (int e = 0; e < NEXP; ++e) { if (acc[e] > b0) { b0 = acc[e]; i0 = e; } }
    float b1 = -3.4e38f; int i1 = 0;
#pragma unroll
    for (int e = 0; e < NEXP; ++e) { if (e != i0 && acc[e] > b1) { b1 = acc[e]; i1 = e; } }
    float p1  = __expf(b1 - b0);           // p0 = 1
    float inv = 1.0f / (1.0f + p1);
    int s0 = atomicAdd(&counts[i0], 1);
    toks[i0 * NTOK + s0] = n; wts[i0 * NTOK + s0] = inv;
    int s1 = atomicAdd(&counts[i1], 1);
    toks[i1 * NTOK + s1] = n; wts[i1 * NTOK + s1] = p1 * inv;
  }
}

// ---------------------------------------------------------------------------
// Fused per-expert MLP: Y[tok] += w * silu(X[tok] @ W1[e]) @ W2[e]
// 32 gathered tokens per WG, full 1024 output cols, DFF in chunks of 64.
// All operands are pre-converted bf16; A frags read straight from global
// (gathered rows), B frags from n-major transposed weights (2 b128 each).
// LDS only re-layouts the SiLU output H (GEMM2's A matrix).
// ---------------------------------------------------------------------------
__global__ __launch_bounds__(256)
void moe_expert_kernel(const unsigned short* __restrict__ xbf,   // [N][D] bf16
                       const unsigned short* __restrict__ w1t,   // [E][DFF][D] bf16
                       const unsigned short* __restrict__ w2t,   // [E][D][DFF] bf16
                       const int* __restrict__ counts,
                       const int* __restrict__ toks, const float* __restrict__ wts,
                       float* __restrict__ y) {
  const int LS = 72;                     // hs row stride (shorts); 144B = 16B-aligned
  int e    = blockIdx.x / TILES_PER_E;
  int tile = blockIdx.x % TILES_PER_E;
  int cnt  = counts[e];
  int t0   = tile * TILE_M;
  if (t0 >= cnt) return;

  __shared__ unsigned short hs[TILE_M * LS];   // H chunk 32x64 bf16
  __shared__ int   stok[TILE_M];
  __shared__ float swt[TILE_M];

  int tid = threadIdx.x;
  if (tid < TILE_M) {
    int idx = t0 + tid;
    if (idx < cnt) { stok[tid] = toks[e * NTOK + idx]; swt[tid] = wts[e * NTOK + idx]; }
    else           { stok[tid] = 0;                    swt[tid] = 0.0f; }
  }
  __syncthreads();

  int wave = tid >> 5;
  int lane = tid & 31;
  int hl   = lane >> 4;
  const unsigned short* w1e = w1t + (size_t)e * DFFDIM * DMODEL;
  const unsigned short* w2e = w2t + (size_t)e * DMODEL * DFFDIM;

  int m1  = (wave & 1) * 16;             // this wave's 16-row half of the token tile
  int nf1 = (wave >> 1) * 16;            // GEMM1: 16-col tile inside the DFF chunk
  int g2  = (wave >> 1) * 256;           // GEMM2: 256-col output group

  // Per-lane A row pointers are loop-invariant (row = m1 + lane%16).
  const unsigned short* xrow = xbf + (size_t)stok[m1 + (lane & 15)] * DMODEL;
  const unsigned short* hrow = hs + (size_t)(m1 + (lane & 15)) * LS;

  const v8f vzero = {0.f, 0.f, 0.f, 0.f, 0.f, 0.f, 0.f, 0.f};
  v8f yacc[16];                          // 32x1024 output tile: 16 C-frags per wave
#pragma unroll
  for (int t = 0; t < 16; ++t) yacc[t] = vzero;

  for (int nf = 0; nf < DFFDIM; nf += NFCH) {
    // ---- GEMM1: hacc(16x16) = X(16x1024) @ W1[:, nf+nf1 .. +16], no barriers ----
    v8f hacc = vzero;
#pragma unroll 4
    for (int k0 = 0; k0 < DMODEL; k0 += 32) {
      v16bf a = load_A_row(xrow, k0);
      v16bf b = load_B_nmajor(w1e, DMODEL, k0, nf + nf1);
      hacc = wmma_bf16(a, b, hacc);
    }

    // ---- SiLU on C-fragment, re-layout through LDS as GEMM2's A ----
#pragma unroll
    for (int v = 0; v < 8; ++v) {
      int row = m1 + 8 * hl + v;         // C layout: M = vgpr + 8*(lane>=16)
      int col = nf1 + (lane & 15);
      float hv = hacc[v];
      hs[row * LS + col] = f32_to_bf16(hv / (1.0f + __expf(-hv)));
    }
    __syncthreads();

    v16bf ha0 = load_A_row(hrow, 0);     // 2x ds_load_b128 each
    v16bf ha1 = load_A_row(hrow, 32);
    if (nf + NFCH < DFFDIM)
      __builtin_prefetch(w2e + (size_t)g2 * DFFDIM + nf + NFCH, 0, 0);

    // ---- GEMM2: Y(32x1024) += H(32x64) @ W2[nf..nf+64, :] ----
#pragma unroll
    for (int t = 0; t < 16; ++t) {
      v16bf wb0 = load_B_nmajor(w2e, DFFDIM, nf,      g2 + t * 16);
      v16bf wb1 = load_B_nmajor(w2e, DFFDIM, nf + 32, g2 + t * 16);
      yacc[t] = wmma_bf16(ha0, wb0, yacc[t]);
      yacc[t] = wmma_bf16(ha1, wb1, yacc[t]);
    }
    __syncthreads();                     // hs rewritten next chunk
  }

  // Weighted scatter-add epilogue; padded rows carry weight 0.
#pragma unroll
  for (int t = 0; t < 16; ++t) {
#pragma unroll
    for (int v = 0; v < 8; ++v) {
      int r   = m1 + 8 * hl + v;
      int col = g2 + t * 16 + (lane & 15);
      atomicAdd(&y[(size_t)stok[r] * DMODEL + col], swt[r] * yacc[t][v]);
    }
  }
}

extern "C" void kernel_launch(void* const* d_in, const int* in_sizes, int n_in,
                              void* d_out, int out_size, void* d_ws, size_t ws_size,
                              hipStream_t stream) {
  const float* x  = (const float*)d_in[0];   // [N, D]
  const float* gw = (const float*)d_in[1];   // [D, E]
  const float* w1 = (const float*)d_in[2];   // [E, D, DFF]
  const float* w2 = (const float*)d_in[3];   // [E, DFF, D]
  float* y = (float*)d_out;                  // [N, D] fp32

  // Workspace layout (256B-aligned segments):
  // counts | token lists | weight lists | x bf16 | W1^T bf16 | W2^T bf16
  char* ws = (char*)d_ws;
  size_t off = 0;
  int*   counts = (int*)(ws + off);            off += 256;
  int*   toks   = (int*)(ws + off);            off += (size_t)NEXP * NTOK * sizeof(int);
  float* wts    = (float*)(ws + off);          off += (size_t)NEXP * NTOK * sizeof(float);
  unsigned short* xbf = (unsigned short*)(ws + off); off += (size_t)NTOK * DMODEL * 2;
  unsigned short* w1t = (unsigned short*)(ws + off); off += (size_t)NEXP * DMODEL * DFFDIM * 2;
  unsigned short* w2t = (unsigned short*)(ws + off); off += (size_t)NEXP * DFFDIM * DMODEL * 2;

  hipMemsetAsync(d_ws, 0, 256, stream);                                // zero counts
  hipMemsetAsync(d_out, 0, (size_t)out_size * sizeof(float), stream);  // y = 0

  // One-time precision/layout pre-pass.
  convert_bf16_kernel<<<(NTOK * DMODEL) / (256 * 8), 256, 0, stream>>>(x, xbf);
  transpose_bf16_kernel<<<dim3(DFFDIM / 32, DMODEL / 32, NEXP), 256, 0, stream>>>(
      w1, w1t, DMODEL, DFFDIM);   // [D][DFF] -> [DFF][D]
  transpose_bf16_kernel<<<dim3(DMODEL / 32, DFFDIM / 32, NEXP), 256, 0, stream>>>(
      w2, w2t, DFFDIM, DMODEL);   // [DFF][D] -> [D][DFF]

  moe_router_kernel<<<NTOK / 8, 256, 0, stream>>>(x, gw, counts, toks, wts);
  moe_expert_kernel<<<NEXP * TILES_PER_E, 256, 0, stream>>>(xbf, w1t, w2t,
                                                            counts, toks, wts, y);
}
